// MLA_14293651161738
// MI455X (gfx1250) — compile-verified, matrix operand order checked by reading, not run on hardware
//
#include <hip/hip_runtime.h>
#include <hip/hip_bf16.h>

typedef __bf16 bf16_t;
typedef __attribute__((ext_vector_type(16))) __bf16 v16bf;
typedef __attribute__((ext_vector_type(8)))  float  v8f;

#if defined(__AMDGCN__) && __has_builtin(__builtin_amdgcn_tensor_load_to_lds)
#define HAVE_TDM 1
#else
#define HAVE_TDM 0
#endif

#if HAVE_TDM
typedef __attribute__((ext_vector_type(4))) unsigned int v4u_t;
typedef __attribute__((ext_vector_type(8))) int v8i_t;
typedef __attribute__((ext_vector_type(4))) int v4i_t;
#endif

#define B_    4
#define S_    2048
#define DIM_  2048
#define H_    16
#define NOPE_ 128
#define ROPE_ 64
#define VD_   128
#define QKD_  192
#define KVR_  512
#define CAT_  576
#define SCALE_ 0.07216878364870322f   /* 192^-0.5 */

// ---------------------------------------------------------------------------
// bf16 A/B fragment load for v_wmma_f32_16x16x32_bf16 (ISA 7.12.2):
// lane L holds row (L&15); K chunks: [8*half .. 8*half+7] and [+16 .. +23].
// Two contiguous 8-elem (16B) chunks -> two ds_load_b128.
// ---------------------------------------------------------------------------
static __device__ __forceinline__ v16bf frag_ld(const bf16_t* base, int row,
                                                int ld, int k0, int lane) {
  const int half = (lane >> 4) & 1;
  const bf16_t* p = base + (size_t)row * ld + k0 + 8 * half;
  union { v16bf v; uint4 u[2]; } r;
  r.u[0] = *(const uint4*)(p);
  r.u[1] = *(const uint4*)(p + 16);
  return r.v;
}

static __device__ __forceinline__ v8f wmma_bf16(v16bf a, v16bf b, v8f c) {
  return __builtin_amdgcn_wmma_f32_16x16x32_bf16(false, a, false, b,
                                                 (short)0, c, false, false);
}

#if HAVE_TDM
// ---------------------------------------------------------------------------
// Tensor Data Mover: 2D tile (h rows of w bf16 elems, row stride in elems)
// from global -> LDS. Optional LDS padding: +4 dwords after every 32 dwords
// (i.e. 64-elem rows land on an ld=72 layout). D# per CDNA5 ISA.
// Issue from one wave; completion via s_wait_tensorcnt then barrier.
// ---------------------------------------------------------------------------
static __device__ __forceinline__ void tdm_load_2d(const void* gptr,
                                                   void* lptr, unsigned w,
                                                   unsigned h,
                                                   unsigned stride_elems,
                                                   bool pad) {
  unsigned long long ga = (unsigned long long)(uintptr_t)gptr;
  v4u_t g0;
  g0[0] = 1u;                                   // count=1 (valid descriptor)
  g0[1] = (unsigned)(uintptr_t)lptr;            // lds_addr (low 32 bits)
  g0[2] = (unsigned)ga;                         // global_addr[31:0]
  g0[3] = (unsigned)((ga >> 32) & 0x01ffffffu) | (2u << 30);  // addr | type=2
  unsigned f = 0x00010000u;                     // data_size=1 -> 2 bytes
  if (pad) f |= (1u << 20) | (4u << 22) | (3u << 25);  // pad 4 dw / 32 dw
  v8i_t g1;
  g1[0] = (int)f;
  g1[1] = (int)(w << 16);                       // tensor_dim0[15:0]
  g1[2] = (int)((w >> 16) | (h << 16));         // dim0[31:16] | dim1[15:0]
  g1[3] = (int)((h >> 16) | (w << 16));         // dim1[31:16] | tile_dim0
  g1[4] = (int)h;                               // tile_dim1 (tile_dim2=0)
  g1[5] = (int)stride_elems;                    // tensor_dim0_stride[31:0]
  g1[6] = 0;
  g1[7] = 0;
  v4i_t z4 = {0, 0, 0, 0};
#if defined(__clang_major__) && (__clang_major__ >= 23)
  v8i_t z8 = {0, 0, 0, 0, 0, 0, 0, 0};
  __builtin_amdgcn_tensor_load_to_lds(g0, g1, z4, z4, z8, 0);
#else
  __builtin_amdgcn_tensor_load_to_lds(g0, g1, z4, z4, 0);
#endif
}
#endif  // HAVE_TDM

// ---------------------------------------------------------------------------
// f32 -> bf16 conversion (grid-stride)
// ---------------------------------------------------------------------------
__global__ void k_cvt(const float* __restrict__ in, bf16_t* __restrict__ out,
                      int n) {
  for (int i = blockIdx.x * blockDim.x + threadIdx.x; i < n;
       i += gridDim.x * blockDim.x)
    out[i] = (bf16_t)in[i];
}

// ---------------------------------------------------------------------------
// Generic bf16 WMMA GEMM: C[z] = A[z](MxK) * B[z](KxN)
// Block = 256 threads (8 waves), tile 256x64, BK=64, wave tile 64x32
// (4x2 WMMA tiles -> 12 ds_load_b128 per 8 WMMA). A tile (and B tile when
// TRANSB) staged by the Tensor Data Mover with hardware LDS padding.
// TRANSB: B stored N-major (N x K row-major). Batched over blockIdx.z.
// ---------------------------------------------------------------------------
template <bool TRANSB, bool OUTBF16>
__global__ __launch_bounds__(256) void k_gemm(
    const bf16_t* __restrict__ A, const bf16_t* __restrict__ Bm,
    void* __restrict__ Cv, int M, int N, int K, int lda, int ldb, int ldc,
    long long sA, long long sB, long long sC) {
  (void)M; (void)N;
  __shared__ __align__(16) bf16_t As[256 * 72];  // ld=72 (pad 8): 36 KB
  __shared__ __align__(16) bf16_t Bs[64 * 72];   // stored [n][k]:  9 KB
  const int tid  = threadIdx.x;
  const int lane = tid & 31;
  const int wv   = tid >> 5;
  const int wm   = wv >> 1;     // 0..3 : 64-row slice
  const int wn   = wv & 1;      // 0..1 : 32-col slice
  const int m0   = blockIdx.y * 256;
  const int n0   = blockIdx.x * 64;
  const bf16_t* Ag = A  + (long long)blockIdx.z * sA + (size_t)m0 * lda;
  const bf16_t* Bg = Bm + (long long)blockIdx.z * sB;

  v8f acc[4][2] = {};
  for (int kt = 0; kt < K; kt += 64) {
    __syncthreads();
#if HAVE_TDM
    if (tid < 32) {
      tdm_load_2d(Ag + kt, As, 64, 256, (unsigned)lda, true);
      if (TRANSB)
        tdm_load_2d(Bg + (size_t)n0 * ldb + kt, Bs, 64, 64, (unsigned)ldb,
                    true);
    }
    if (!TRANSB) {  // K x N row-major: transpose into [n][k] (no TDM transpose)
      for (int e = tid; e < 64 * 64; e += 256) {
        int n = e & 63, k = e >> 6;
        Bs[n * 72 + k] = Bg[(size_t)(kt + k) * ldb + n0 + n];
      }
    }
    if (tid < 32) __builtin_amdgcn_s_wait_tensorcnt((short)0);
#else
    for (int u = tid; u < 256 * 32; u += 256) {
      int row = u >> 5, i = u & 31;
      ((unsigned int*)As)[row * 36 + i] =
          *(const unsigned int*)(Ag + (size_t)row * lda + kt + 2 * i);
    }
    if (TRANSB) {
      for (int u = tid; u < 64 * 32; u += 256) {
        int n = u >> 5, i = u & 31;
        ((unsigned int*)Bs)[n * 36 + i] =
            *(const unsigned int*)(Bg + (size_t)(n0 + n) * ldb + kt + 2 * i);
      }
    } else {
      for (int e = tid; e < 64 * 64; e += 256) {
        int n = e & 63, k = e >> 6;
        Bs[n * 72 + k] = Bg[(size_t)(kt + k) * ldb + n0 + n];
      }
    }
#endif
    __syncthreads();
#pragma unroll
    for (int ks = 0; ks < 64; ks += 32) {
      v16bf b0 = frag_ld(Bs, wn * 32 + (lane & 15),      72, ks, lane);
      v16bf b1 = frag_ld(Bs, wn * 32 + 16 + (lane & 15), 72, ks, lane);
#pragma unroll
      for (int mi = 0; mi < 4; ++mi) {
        v16bf a = frag_ld(As, wm * 64 + mi * 16 + (lane & 15), 72, ks, lane);
        acc[mi][0] = wmma_bf16(a, b0, acc[mi][0]);
        acc[mi][1] = wmma_bf16(a, b1, acc[mi][1]);
      }
    }
  }
  // C/D layout: lane L -> col (L&15); VGPR r -> row r (lanes 0-15) / r+8
  const int half = lane >> 4;
#pragma unroll
  for (int mi = 0; mi < 4; ++mi)
#pragma unroll
    for (int ni = 0; ni < 2; ++ni)
#pragma unroll
      for (int r = 0; r < 8; ++r) {
        size_t row = (size_t)m0 + wm * 64 + mi * 16 + r + 8 * half;
        size_t col = (size_t)n0 + wn * 32 + ni * 16 + (lane & 15);
        size_t off = (size_t)((long long)blockIdx.z * sC) + row * ldc + col;
        if (OUTBF16) ((bf16_t*)Cv)[off] = (bf16_t)acc[mi][ni][r];
        else         ((float*)Cv)[off]  = acc[mi][ni][r];
      }
}

// ---------------------------------------------------------------------------
// RoPE on q_pe (last 64 of each head's 192) -> q_cat[:, 512:576]
// ---------------------------------------------------------------------------
__global__ void k_ropeq(const bf16_t* __restrict__ q,
                        const float* __restrict__ fcos,
                        const float* __restrict__ fsin,
                        bf16_t* __restrict__ qcat) {
  int idx = blockIdx.x * blockDim.x + threadIdx.x;
  if (idx >= B_ * S_ * H_ * 32) return;
  int i = idx & 31;
  int h = (idx >> 5) & (H_ - 1);
  int g = idx >> 9;           // flattened (b,s)
  int s = g & (S_ - 1);
  const bf16_t* src = q + (size_t)g * (H_ * QKD_) + h * QKD_ + NOPE_;
  float xr = (float)src[2 * i];
  float xi = (float)src[2 * i + 1];
  float c = fcos[s * 32 + i], sn = fsin[s * 32 + i];
  bf16_t* dst = qcat + ((size_t)g * H_ + h) * CAT_ + KVR_;
  dst[2 * i]     = (bf16_t)(xr * c - xi * sn);
  dst[2 * i + 1] = (bf16_t)(xr * sn + xi * c);
}

// ---------------------------------------------------------------------------
// RMSNorm(kv latent) + RoPE(k_pe) -> k_cat (bf16, 576 wide). One block/row.
// ---------------------------------------------------------------------------
__global__ __launch_bounds__(128) void k_kvprep(
    const float* __restrict__ kvfull, const float* __restrict__ kw,
    const float* __restrict__ fcos, const float* __restrict__ fsin,
    bf16_t* __restrict__ kcat) {
  __shared__ float red[4];
  int g = blockIdx.x;
  int tid = threadIdx.x;
  const float* row = kvfull + (size_t)g * CAT_;
  float ss = 0.f;
  for (int c = tid; c < KVR_; c += 128) { float v = row[c]; ss += v * v; }
  for (int off = 16; off >= 1; off >>= 1) ss += __shfl_xor(ss, off);
  if ((tid & 31) == 0) red[tid >> 5] = ss;
  __syncthreads();
  ss = red[0] + red[1] + red[2] + red[3];
  float rn = rsqrtf(ss * (1.0f / KVR_) + 1e-6f);
  bf16_t* out = kcat + (size_t)g * CAT_;
  for (int c = tid; c < KVR_; c += 128) out[c] = (bf16_t)(row[c] * rn * kw[c]);
  if (tid < 32) {
    int i = tid;
    int s = g & (S_ - 1);
    float xr = row[KVR_ + 2 * i], xi = row[KVR_ + 2 * i + 1];
    float c = fcos[s * 32 + i], sn = fsin[s * 32 + i];
    out[KVR_ + 2 * i]     = (bf16_t)(xr * c - xi * sn);
    out[KVR_ + 2 * i + 1] = (bf16_t)(xr * sn + xi * c);
  }
}

// ---------------------------------------------------------------------------
// Flash attention over the 512-dim latent (+64 rope) with causal mask.
// Block = 128 threads (4 waves), 16 queries, K-chunks of 32 staged in LDS
// (via TDM when available). Each wave owns a 128-wide KVR output slice.
// ---------------------------------------------------------------------------
__global__ __launch_bounds__(128) void k_attn(const bf16_t* __restrict__ qcat,
                                              const bf16_t* __restrict__ kcat,
                                              bf16_t* __restrict__ olat) {
  __shared__ __align__(16) bf16_t q_s[16 * CAT_];   // 18 KB
  __shared__ __align__(16) bf16_t k_s[32 * CAT_];   // 36 KB
  __shared__ __align__(16) bf16_t p_s[4][16 * 32];  // 4 KB
  const int tid  = threadIdx.x;
  const int lane = tid & 31;
  const int w    = tid >> 5;
  const int half = lane >> 4;
  const int qb   = blockIdx.x;
  const int h    = blockIdx.y;
  const int b    = blockIdx.z;
  const size_t qbase = (size_t)b * S_ + qb * 16;

  // stage 16 query rows of q_cat (strided rows: one per (b,s), this head)
#if HAVE_TDM
  if (tid < 32) {
    tdm_load_2d(qcat + (qbase * H_ + h) * CAT_, q_s, CAT_, 16,
                (unsigned)(H_ * CAT_), false);
    __builtin_amdgcn_s_wait_tensorcnt((short)0);
  }
#else
  for (int u = tid; u < 16 * 288; u += 128) {
    int row = u / 288, i = u % 288;
    ((unsigned int*)q_s)[row * 288 + i] =
        *(const unsigned int*)(qcat + ((qbase + row) * H_ + h) * CAT_ + 2 * i);
  }
#endif

  v8f acc[8] = {};
  float rmax[8], rsum[8];
#pragma unroll
  for (int r = 0; r < 8; ++r) { rmax[r] = -3.0e38f; rsum[r] = 0.f; }

  const int tend = qb * 16 + 16;  // causal: only chunks with t0 < tend
  for (int t0 = 0; t0 < tend; t0 += 32) {
    __syncthreads();
#if HAVE_TDM
    if (tid < 32) {
      tdm_load_2d(kcat + ((size_t)b * S_ + t0) * CAT_, k_s, CAT_, 32,
                  (unsigned)CAT_, false);
      __builtin_amdgcn_s_wait_tensorcnt((short)0);
    }
#else
    for (int u = tid; u < 32 * 288; u += 128) {
      int row = u / 288, i = u % 288;
      ((unsigned int*)k_s)[row * 288 + i] =
          *(const unsigned int*)(kcat +
                                 ((size_t)b * S_ + t0 + row) * CAT_ + 2 * i);
    }
#endif
    __syncthreads();

    // scores 16x32: contraction over 576 (latent + rope in one pass)
    v8f sc[2] = {};
#pragma unroll
    for (int kk = 0; kk < 18; ++kk) {
      v16bf aq = frag_ld(q_s, lane & 15,        CAT_, kk * 32, lane);
      v16bf b0 = frag_ld(k_s, (lane & 15),      CAT_, kk * 32, lane);
      v16bf b1 = frag_ld(k_s, 16 + (lane & 15), CAT_, kk * 32, lane);
      sc[0] = wmma_bf16(aq, b0, sc[0]);
      sc[1] = wmma_bf16(aq, b1, sc[1]);
    }

    // scale + causal mask + online softmax (row = r + 8*half per C layout)
#pragma unroll
    for (int r = 0; r < 8; ++r) {
      int qrow = qb * 16 + r + 8 * half;
      float s0 = sc[0][r] * SCALE_;
      float s1 = sc[1][r] * SCALE_;
      int ta = t0 + (lane & 15);
      if (ta > qrow)      s0 = -1.0e9f;
      if (ta + 16 > qrow) s1 = -1.0e9f;
      float mx = fmaxf(s0, s1);
      mx = fmaxf(mx, __shfl_xor(mx, 1));
      mx = fmaxf(mx, __shfl_xor(mx, 2));
      mx = fmaxf(mx, __shfl_xor(mx, 4));
      mx = fmaxf(mx, __shfl_xor(mx, 8));
      float mnew = fmaxf(rmax[r], mx);
      float corr = __expf(rmax[r] - mnew);
      rmax[r] = mnew;
      float p0 = __expf(s0 - mnew);
      float p1 = __expf(s1 - mnew);
      float ps = p0 + p1;
      ps += __shfl_xor(ps, 1);
      ps += __shfl_xor(ps, 2);
      ps += __shfl_xor(ps, 4);
      ps += __shfl_xor(ps, 8);
      rsum[r] = rsum[r] * corr + ps;
#pragma unroll
      for (int n = 0; n < 8; ++n) acc[n][r] = acc[n][r] * corr;
      int prow = r + 8 * half;
      p_s[w][prow * 32 + (lane & 15)]      = (bf16_t)p0;
      p_s[w][prow * 32 + 16 + (lane & 15)] = (bf16_t)p1;
    }
    __syncthreads();

    // O += P(16x32) @ kv_c(32 x 128 slice owned by this wave)
    v16bf pa = frag_ld(p_s[w], lane & 15, 32, 0, lane);
#pragma unroll
    for (int n = 0; n < 8; ++n) {
      int c = w * 128 + n * 16 + (lane & 15);
      union { v16bf v; bf16_t e[16]; } bb;
#pragma unroll
      for (int i = 0; i < 8; ++i) {
        bb.e[i]     = k_s[(8 * half + i) * CAT_ + c];       // K = 8h..8h+7
        bb.e[i + 8] = k_s[(8 * half + 16 + i) * CAT_ + c];  // K = 8h+16..+23
      }
      acc[n] = wmma_bf16(pa, bb.v, acc[n]);
    }
  }

  // normalize and store o_lat (bf16)
#pragma unroll
  for (int r = 0; r < 8; ++r) {
    float inv = 1.0f / rsum[r];
    size_t grow = qbase + r + 8 * half;
    size_t base = (grow * H_ + h) * KVR_ + w * 128 + (lane & 15);
#pragma unroll
    for (int n = 0; n < 8; ++n)
      olat[base + n * 16] = (bf16_t)(acc[n][r] * inv);
  }
}

// ---------------------------------------------------------------------------
// Host-side orchestration
// ---------------------------------------------------------------------------
extern "C" void kernel_launch(void* const* d_in, const int* in_sizes, int n_in,
                              void* d_out, int out_size, void* d_ws,
                              size_t ws_size, hipStream_t stream) {
  (void)in_sizes; (void)n_in; (void)out_size; (void)ws_size;
  const float* x    = (const float*)d_in[0];
  const float* wq   = (const float*)d_in[1];
  const float* wkva = (const float*)d_in[2];
  const float* kvw  = (const float*)d_in[3];
  const float* wkvb = (const float*)d_in[4];
  const float* wo   = (const float*)d_in[5];
  const float* fcos = (const float*)d_in[6];
  const float* fsin = (const float*)d_in[7];
  // d_in[8] = mask: causal, applied analytically in k_attn
  float* out = (float*)d_out;

  char* p = (char*)d_ws;
  auto take = [&](size_t bytes) -> char* {
    char* r = p;
    p += (bytes + 255) & ~(size_t)255;
    return r;
  };
  const size_t M = (size_t)B_ * S_;  // 8192
  bf16_t* xb    = (bf16_t*)take(M * DIM_ * 2);
  bf16_t* wqb   = (bf16_t*)take((size_t)DIM_ * (H_ * QKD_) * 2);
  bf16_t* wkvab = (bf16_t*)take((size_t)DIM_ * CAT_ * 2);
  bf16_t* wkvbb = (bf16_t*)take((size_t)(H_ * (NOPE_ + VD_)) * KVR_ * 2);
  bf16_t* wob   = (bf16_t*)take((size_t)(H_ * VD_) * DIM_ * 2);
  bf16_t* qbf   = (bf16_t*)take(M * (H_ * QKD_) * 2);
  float*  kvf   = (float*) take(M * CAT_ * 4);
  bf16_t* qcat  = (bf16_t*)take(M * H_ * CAT_ * 2);
  bf16_t* kcat  = (bf16_t*)take(M * CAT_ * 2);
  bf16_t* olat  = (bf16_t*)take(M * H_ * KVR_ * 2);
  bf16_t* ohead = (bf16_t*)take(M * (H_ * VD_) * 2);

  // fp32 -> bf16 casts
  k_cvt<<<4096, 256, 0, stream>>>(x, xb, (int)(M * DIM_));
  k_cvt<<<2048, 256, 0, stream>>>(wq, wqb, DIM_ * H_ * QKD_);
  k_cvt<<<512,  256, 0, stream>>>(wkva, wkvab, DIM_ * CAT_);
  k_cvt<<<512,  256, 0, stream>>>(wkvb, wkvbb, H_ * (NOPE_ + VD_) * KVR_);
  k_cvt<<<1024, 256, 0, stream>>>(wo, wob, DIM_ * DIM_);

  // q = x @ wq : (8192 x 3072), K=2048 -> bf16
  k_gemm<false, true><<<dim3(3072 / 64, 8192 / 256, 1), 256, 0, stream>>>(
      xb, wqb, qbf, 8192, 3072, 2048, 2048, 3072, 3072, 0, 0, 0);
  // kv_full = x @ wkv_a : (8192 x 576), K=2048 -> f32
  k_gemm<false, false><<<dim3(576 / 64, 8192 / 256, 1), 256, 0, stream>>>(
      xb, wkvab, kvf, 8192, 576, 2048, 2048, 576, 576, 0, 0, 0);
  // RMSNorm + k rope -> k_cat
  k_kvprep<<<(int)M, 128, 0, stream>>>(kvf, kvw, fcos, fsin, kcat);
  // absorbed q: per head, q_nope(8192x128) @ w_b[h,:128,:](128x512) -> qcat[:, :512]
  k_gemm<false, true><<<dim3(512 / 64, 8192 / 256, H_), 256, 0, stream>>>(
      qbf, wkvbb, qcat, 8192, 512, 128, 3072, 512, H_ * CAT_,
      (long long)QKD_, (long long)(NOPE_ + VD_) * KVR_, (long long)CAT_);
  // q rope -> qcat[:, 512:]
  k_ropeq<<<(B_ * S_ * H_ * 32) / 256, 256, 0, stream>>>(qbf, fcos, fsin, qcat);
  // flash attention -> o_lat (B*S, H, 512)
  k_attn<<<dim3(S_ / 16, H_, B_), 128, 0, stream>>>(qcat, kcat, olat);
  // latent -> head dim: per head, o_lat(8192x512) @ w_b[h,128:,:]^T(512x128)
  k_gemm<true, true><<<dim3(128 / 64, 8192 / 256, H_), 256, 0, stream>>>(
      olat, wkvbb + (size_t)NOPE_ * KVR_, ohead, 8192, 128, 512,
      H_ * KVR_, 512, H_ * VD_,
      (long long)KVR_, (long long)(NOPE_ + VD_) * KVR_, (long long)VD_);
  // out = o_heads(8192x2048) @ wo(2048x2048) -> f32
  k_gemm<false, false><<<dim3(2048 / 64, 8192 / 256, 1), 256, 0, stream>>>(
      ohead, wob, out, 8192, 2048, 2048, 2048, 2048, 2048, 0, 0, 0);
}
